// GATLayer_65206193488366
// MI455X (gfx1250) — matrix operand order repaired and checked by reading, not compiled, not run on hardware
//
#include <hip/hip_runtime.h>
#include <stdint.h>

typedef __attribute__((ext_vector_type(16))) __bf16        v16bf;
typedef __attribute__((ext_vector_type(8)))  float         v8f;
typedef __attribute__((ext_vector_type(8)))  unsigned int  v8u;

#define HEADS     8
#define DHEAD     64
#define HD        512          // H*D == IN_DIM
#define NEG_SLOPE 0.2f
#define LN_EPS    1e-5f

// ---------------------------------------------------------------- helpers

__device__ __forceinline__ unsigned short f2bf(float f) {
  unsigned u = __float_as_uint(f);
  u += 0x7FFFu + ((u >> 16) & 1u);      // round-to-nearest-even
  return (unsigned short)(u >> 16);
}

__global__ void k_f32_to_bf16(const float* __restrict__ in,
                              unsigned short* __restrict__ out, long long n) {
  long long i = (long long)blockIdx.x * blockDim.x + threadIdx.x;
  long long s = (long long)gridDim.x * blockDim.x;
  for (; i < n; i += s) out[i] = f2bf(in[i]);
}

__global__ void k_zero_f32(float* __restrict__ p, long long n) {
  long long i = (long long)blockIdx.x * blockDim.x + threadIdx.x;
  long long s = (long long)gridDim.x * blockDim.x;
  for (; i < n; i += s) p[i] = 0.0f;
}

// ---------------------------------------------------------------- GEMM (WMMA bf16)
// Hm[M, 512] = Xb[M, 512] * Wb[512, 512]^T     (both operands K-major: "NT" gemm)
// One wave computes a 16x64 strip of C (4 accumulators, A fragment reused 4x).
// Register double-buffered over K: loads for step k+32 are issued before the
// WMMAs of step k, so s_wait_loadcnt can retire with later loads in flight.

__global__ __launch_bounds__(256)
void k_gemm_wmma_bf16(const unsigned short* __restrict__ Xb,
                      const unsigned short* __restrict__ Wb,
                      float* __restrict__ Hm, int M) {
  const int K = HD;
  int wave = (blockIdx.x * blockDim.x + threadIdx.x) >> 5;
  int lane = threadIdx.x & 31;
  const int tilesN = HD / 64;               // 8 column strips of 64
  int tm = wave / tilesN;
  int tn = wave % tilesN;
  int m0 = tm * 16;
  if (m0 >= M) return;
  int n0 = tn * 64;

  int grp = lane >> 4;                      // lane group 0/1
  int l15 = lane & 15;
  int arow = m0 + l15;                      // A: lane -> M row
  if (arow >= M) arow = M - 1;              // tail clamp (store is guarded)

  // A: lane-group K offset folded into base; element v at ((v&3)<<1)+((v>>2)<<4)
  const unsigned short* aptr  = Xb + (size_t)arow * K + (grp << 3);
  // B: lane = N column, lane-group K offset folded into base; element v at ((v&3)<<1)+((v>>2)<<4)
  const unsigned short* bbase = Wb + (size_t)(n0 + l15) * K + (grp << 3);

  v8u au[2];
  v8u bu[2][4];

  auto load_stage = [&](int k0, int s) {
#pragma unroll
    for (int v = 0; v < 8; ++v) {
      int k = ((v & 3) << 1) + ((v >> 2) << 4);
      au[s][v] = *(const unsigned int*)(aptr + k0 + k);
    }
#pragma unroll
    for (int j = 0; j < 4; ++j) {
#pragma unroll
      for (int v = 0; v < 8; ++v) {
        bu[s][j][v] = *(const unsigned int*)(bbase + (size_t)j * 16 * K + k0 + ((v & 3) << 1) + ((v >> 2) << 4));
      }
    }
  };

  v8f acc[4] = { {}, {}, {}, {} };
  load_stage(0, 0);

#pragma unroll 1
  for (int k0 = 0; k0 < K; k0 += 64) {
    load_stage(k0 + 32, 1);                 // prefetch stage 1 while stage 0 computes
    {
      v16bf a = __builtin_bit_cast(v16bf, au[0]);
#pragma unroll
      for (int j = 0; j < 4; ++j)
        acc[j] = __builtin_amdgcn_wmma_f32_16x16x32_bf16(
            false, a, false, __builtin_bit_cast(v16bf, bu[0][j]),
            (short)0, acc[j], false, false);
    }
    if (k0 + 64 < K) load_stage(k0 + 64, 0); // prefetch stage 0 while stage 1 computes
    {
      v16bf a = __builtin_bit_cast(v16bf, au[1]);
#pragma unroll
      for (int j = 0; j < 4; ++j)
        acc[j] = __builtin_amdgcn_wmma_f32_16x16x32_bf16(
            false, a, false, __builtin_bit_cast(v16bf, bu[1][j]),
            (short)0, acc[j], false, false);
    }
  }

  // ---- store C: lane&15 = N, VGPR v -> M = v + 8*grp
  int mbase = m0 + (grp << 3);
  float* outp = Hm + (size_t)mbase * HD + n0 + l15;
#pragma unroll
  for (int j = 0; j < 4; ++j) {
#pragma unroll
    for (int v = 0; v < 8; ++v) {
      if (mbase + v < M) outp[(size_t)v * HD + j * 16] = acc[j][v];
    }
  }
}

// ---------------------------------------------------------------- per-node scores

__global__ void k_node_scores(const float* __restrict__ Hm,
                              const float* __restrict__ a_src,
                              const float* __restrict__ a_dst,
                              float* __restrict__ ssrc,
                              float* __restrict__ sdst, int N) {
  int total = N * HEADS;
  int stride = gridDim.x * blockDim.x;
  for (int idx = blockIdx.x * blockDim.x + threadIdx.x; idx < total; idx += stride) {
    int n = idx >> 3, h = idx & 7;
    const float4* hp = (const float4*)(Hm + (size_t)n * HD + h * DHEAD);
    const float4* as = (const float4*)(a_src + h * DHEAD);
    const float4* ad = (const float4*)(a_dst + h * DHEAD);
    float ss = 0.f, sd = 0.f;
#pragma unroll
    for (int i = 0; i < DHEAD / 4; ++i) {
      float4 hv = hp[i], av = as[i], dv = ad[i];
      ss += hv.x * av.x + hv.y * av.y + hv.z * av.z + hv.w * av.w;
      sd += hv.x * dv.x + hv.y * dv.y + hv.z * dv.z + hv.w * dv.w;
    }
    ssrc[idx] = ss;
    sdst[idx] = sd;
  }
}

// ---------------------------------------------------------------- edge pass 1: logits + segment max
// e_max init 0 and only positives can win (reference includes 0 in max), so
// int-bits atomicMax on positive floats is exact.

__global__ void k_edge_logits(const int* __restrict__ src, const int* __restrict__ dst,
                              const int* __restrict__ etype, const float* __restrict__ ew,
                              const float* __restrict__ ssrc, const float* __restrict__ sdst,
                              const float* __restrict__ et_emb,
                              float* __restrict__ ebuf, int* __restrict__ emax_bits,
                              long long E) {
  long long total = E * HEADS;
  long long stride = (long long)gridDim.x * blockDim.x;
  for (long long i = (long long)blockIdx.x * blockDim.x + threadIdx.x; i < total; i += stride) {
    long long e = i >> 3;
    int h = (int)(i & 7);
    int s = src[e], d = dst[e];
    float v = ssrc[(size_t)s * HEADS + h] + sdst[(size_t)d * HEADS + h];
    v = v > 0.f ? v : v * NEG_SLOPE;
    v *= (1.f + et_emb[etype[e] * HEADS + h]);
    v *= ew[e];
    ebuf[i] = v;
    if (v > 0.f) atomicMax(&emax_bits[(size_t)d * HEADS + h], __float_as_int(v));
  }
}

// ---------------------------------------------------------------- edge pass 2: exp + segment sum

__global__ void k_edge_exp(const int* __restrict__ dst, const float* __restrict__ emax,
                           float* __restrict__ ebuf, float* __restrict__ esum, long long E) {
  long long total = E * HEADS;
  long long stride = (long long)gridDim.x * blockDim.x;
  for (long long i = (long long)blockIdx.x * blockDim.x + threadIdx.x; i < total; i += stride) {
    long long e = i >> 3;
    int h = (int)(i & 7);
    int d = dst[e];
    float ex = __expf(ebuf[i] - emax[(size_t)d * HEADS + h]);
    ebuf[i] = ex;
    atomicAdd(&esum[(size_t)d * HEADS + h], ex);
  }
}

// ---------------------------------------------------------------- edge pass 3: weighted scatter-add
// 512 consecutive threads per edge -> coalesced gather of h[src] row and
// coalesced atomics on out[dst] row (out is L2-resident: 100MB < 192MB L2).

__global__ void k_scatter(const int* __restrict__ src, const int* __restrict__ dst,
                          const float* __restrict__ ebuf, const float* __restrict__ esum,
                          const float* __restrict__ Hm, float* __restrict__ out, long long E) {
  long long total = E << 9;
  long long stride = (long long)gridDim.x * blockDim.x;
  for (long long i = (long long)blockIdx.x * blockDim.x + threadIdx.x; i < total; i += stride) {
    long long e = i >> 9;
    int hd = (int)(i & 511);
    int h = hd >> 6;
    int s = src[e], d = dst[e];
    float alpha = ebuf[e * HEADS + h] / (esum[(size_t)d * HEADS + h] + 1e-10f);
    atomicAdd(&out[(size_t)d * HD + hd], Hm[(size_t)s * HD + hd] * alpha);
  }
}

// ---------------------------------------------------------------- residual + LayerNorm (wave32 per node)

__global__ __launch_bounds__(256)
void k_residual_ln(float* __restrict__ out, const float* __restrict__ x,
                   const float* __restrict__ gamma, const float* __restrict__ beta, int N) {
  int node = (blockIdx.x * blockDim.x + threadIdx.x) >> 5;
  int lane = threadIdx.x & 31;
  if (node >= N) return;
  size_t base = (size_t)node * HD;

  float4 vals[4];
  float sum = 0.f, sq = 0.f;
#pragma unroll
  for (int i = 0; i < 4; ++i) {
    int c = (i * 32 + lane) * 4;
    float4 a = *(const float4*)(out + base + c);
    float4 b = *(const float4*)(x + base + c);
    a.x += b.x; a.y += b.y; a.z += b.z; a.w += b.w;
    vals[i] = a;
    sum += a.x + a.y + a.z + a.w;
    sq  += a.x * a.x + a.y * a.y + a.z * a.z + a.w * a.w;
  }
#pragma unroll
  for (int o = 16; o > 0; o >>= 1) {
    sum += __shfl_xor(sum, o, 32);
    sq  += __shfl_xor(sq, o, 32);
  }
  float mu   = sum * (1.f / HD);
  float var  = sq * (1.f / HD) - mu * mu;
  float rstd = rsqrtf(var + LN_EPS);
#pragma unroll
  for (int i = 0; i < 4; ++i) {
    int c = (i * 32 + lane) * 4;
    float4 g  = *(const float4*)(gamma + c);
    float4 bt = *(const float4*)(beta + c);
    float4 a  = vals[i];
    a.x = (a.x - mu) * rstd * g.x + bt.x;
    a.y = (a.y - mu) * rstd * g.y + bt.y;
    a.z = (a.z - mu) * rstd * g.z + bt.z;
    a.w = (a.w - mu) * rstd * g.w + bt.w;
    *(float4*)(out + base + c) = a;
  }
}

// ---------------------------------------------------------------- launch

extern "C" void kernel_launch(void* const* d_in, const int* in_sizes, int n_in,
                              void* d_out, int out_size, void* d_ws, size_t ws_size,
                              hipStream_t stream) {
  const float* x      = (const float*)d_in[0];
  const int*   eidx   = (const int*)d_in[1];
  const int*   etype  = (const int*)d_in[2];
  const float* ew     = (const float*)d_in[3];
  const float* W      = (const float*)d_in[4];
  const float* a_src  = (const float*)d_in[5];
  const float* a_dst  = (const float*)d_in[6];
  const float* et_emb = (const float*)d_in[7];
  const float* gamma  = (const float*)d_in[8];
  const float* beta   = (const float*)d_in[9];
  float* out = (float*)d_out;

  const long long E = in_sizes[2];          // edge_type count
  const int N = in_sizes[0] / HD;
  const int* src = eidx;                    // edge_index[0,:]
  const int* dst = eidx + E;                // edge_index[1,:]

  // ---- workspace carve-out (256B aligned)
  char* wp = (char*)d_ws;
  auto carve = [&](size_t bytes) -> char* {
    char* p = wp;
    wp += (bytes + 255) & ~(size_t)255;
    return p;
  };
  unsigned short* Xb  = (unsigned short*)carve((size_t)N * HD * 2);
  unsigned short* Wb  = (unsigned short*)carve((size_t)HD * HD * 2);
  float* Hm   = (float*)carve((size_t)N * HD * 4);
  float* ssrc = (float*)carve((size_t)N * HEADS * 4);
  float* sdst = (float*)carve((size_t)N * HEADS * 4);
  float* emax = (float*)carve((size_t)N * HEADS * 4);
  float* esum = (float*)carve((size_t)N * HEADS * 4);
  float* ebuf = (float*)carve((size_t)E * HEADS * 4);

  // 0) f32 -> bf16 operand conversion
  k_f32_to_bf16<<<2048, 256, 0, stream>>>(x, Xb, (long long)N * HD);
  k_f32_to_bf16<<<(HD * HD + 255) / 256, 256, 0, stream>>>(W, Wb, (long long)HD * HD);

  // 1) projection GEMM (WMMA bf16 -> f32)
  {
    int tilesM = (N + 15) / 16;
    int waves  = tilesM * (HD / 64);
    int blocks = (waves + 7) / 8;           // 8 waves (256 threads) per block
    k_gemm_wmma_bf16<<<blocks, 256, 0, stream>>>(Xb, Wb, Hm, N);
  }

  // 2) per-node attention scores
  k_node_scores<<<1024, 256, 0, stream>>>(Hm, a_src, a_dst, ssrc, sdst, N);

  // 3) zero accumulators
  k_zero_f32<<<512, 256, 0, stream>>>(emax, (long long)N * HEADS);
  k_zero_f32<<<512, 256, 0, stream>>>(esum, (long long)N * HEADS);
  k_zero_f32<<<2048, 256, 0, stream>>>(out, (long long)N * HD);

  // 4) edge logits + segment max
  {
    long long t = E * HEADS;
    k_edge_logits<<<(int)((t + 255) / 256), 256, 0, stream>>>(
        src, dst, etype, ew, ssrc, sdst, et_emb, ebuf, (int*)emax, E);
  }

  // 5) exp + segment sum
  {
    long long t = E * HEADS;
    k_edge_exp<<<(int)((t + 255) / 256), 256, 0, stream>>>(dst, emax, ebuf, esum, E);
  }

  // 6) weighted message scatter-add
  {
    long long t = E << 9;
    k_scatter<<<(int)((t + 255) / 256), 256, 0, stream>>>(src, dst, ebuf, esum, Hm, out, E);
  }

  // 7) residual + LayerNorm
  {
    int blocks = (N + 7) / 8;               // one wave32 per node, 8 waves per block
    k_residual_ln<<<blocks, 256, 0, stream>>>(out, x, gamma, beta, N);
  }
}
